// TernaryLinear_18425409700089
// MI455X (gfx1250) — compile-verified, matrix operand order checked by reading, not run on hardware
//
#include <hip/hip_runtime.h>
#include <hip/hip_bf16.h>

// ---------------------------------------------------------------------------
// TernaryLinear (BitNet b1.58) for MI455X / gfx1250
//   y[t,o] = sum_d x[t,d] * ternary(w)[o,d] + b[o]
//   ternary(w) = sign(w) * (|w| > 0.5*mean(|w|))
//
// Compute-bound bf16 WMMA GEMM (1.1 TFLOP vs ~40us HBM traffic).
// Primary path: pre-quantize w -> bf16 {-1,0,1}, convert x -> bf16 into d_ws,
// then GEMM with Tensor Data Mover (TENSOR_LOAD_TO_LDS) double-buffered
// staging pipelined against v_wmma_f32_16x16x32_bf16.
// Fallback (small ws): fused quantize-in-staging kernel (round-1 version).
// ---------------------------------------------------------------------------

#define TOKENS 8192
#define D_IN   4096
#define D_OUT  16384

typedef __attribute__((ext_vector_type(16))) __bf16 v16bf;
typedef __attribute__((ext_vector_type(8)))  __bf16 v8bf;
typedef __attribute__((ext_vector_type(4)))  __bf16 v4bf;
typedef __attribute__((ext_vector_type(8)))  float  v8f;
typedef __attribute__((ext_vector_type(4)))  unsigned int v4u;
typedef __attribute__((ext_vector_type(8)))  int    v8i;
typedef __attribute__((ext_vector_type(4)))  int    v4i;

// ---------------- alpha = mean(|w|), two-pass deterministic reduction -------

__global__ void absmean_partial_kernel(const float* __restrict__ w,
                                       float* __restrict__ partial, int n) {
    __shared__ float red[256];
    float s = 0.0f;
    for (int i = blockIdx.x * blockDim.x + threadIdx.x; i < n;
         i += gridDim.x * blockDim.x) {
        s += fabsf(w[i]);
    }
    red[threadIdx.x] = s;
    __syncthreads();
    for (int off = 128; off > 0; off >>= 1) {
        if (threadIdx.x < (unsigned)off) red[threadIdx.x] += red[threadIdx.x + off];
        __syncthreads();
    }
    if (threadIdx.x == 0) partial[blockIdx.x] = red[0];
}

__global__ void absmean_final_kernel(const float* __restrict__ partial,
                                     float* __restrict__ alpha_out,
                                     int nparts, float inv_n) {
    __shared__ float red[256];
    float s = 0.0f;
    for (int i = threadIdx.x; i < nparts; i += 256) s += partial[i];
    red[threadIdx.x] = s;
    __syncthreads();
    for (int off = 128; off > 0; off >>= 1) {
        if (threadIdx.x < (unsigned)off) red[threadIdx.x] += red[threadIdx.x + off];
        __syncthreads();
    }
    if (threadIdx.x == 0) alpha_out[0] = red[0] * inv_n;
}

// ---------------- pre-pass: w -> ternary bf16, x -> bf16 --------------------

__global__ void quant_w_kernel(const float* __restrict__ w,
                               const float* __restrict__ alpha,
                               __bf16* __restrict__ wq, int n4) {
    const float thr = 0.5f * alpha[0];
    for (int i = blockIdx.x * blockDim.x + threadIdx.x; i < n4;
         i += gridDim.x * blockDim.x) {
        const float4 v = ((const float4*)w)[i];
        v4bf p;
        p[0] = (__bf16)((fabsf(v.x) > thr) ? (v.x > 0.f ? 1.f : -1.f) : 0.f);
        p[1] = (__bf16)((fabsf(v.y) > thr) ? (v.y > 0.f ? 1.f : -1.f) : 0.f);
        p[2] = (__bf16)((fabsf(v.z) > thr) ? (v.z > 0.f ? 1.f : -1.f) : 0.f);
        p[3] = (__bf16)((fabsf(v.w) > thr) ? (v.w > 0.f ? 1.f : -1.f) : 0.f);
        ((v4bf*)wq)[i] = p;
    }
}

__global__ void conv_x_kernel(const float* __restrict__ x,
                              __bf16* __restrict__ xq, int n4) {
    for (int i = blockIdx.x * blockDim.x + threadIdx.x; i < n4;
         i += gridDim.x * blockDim.x) {
        const float4 v = ((const float4*)x)[i];
        v4bf p;
        p[0] = (__bf16)v.x; p[1] = (__bf16)v.y;
        p[2] = (__bf16)v.z; p[3] = (__bf16)v.w;
        ((v4bf*)xq)[i] = p;
    }
}

// ---------------- TDM descriptor helper ------------------------------------
// 2-D tile load: tile_d0 (contiguous, K) x tile_d1 (rows), element = 2 bytes.
// LDS padding: 4 DWORDs (16B) after every 32 DWORDs (128B) -> row stride
// 144 B == 72 halves, matching compute-side addressing. Fields per
// CDNA5 ISA D# Group0/Group1.

#define BM 128
#define BN 128
#define BK 64
#define LDSK (BK + 8)   // 72 halves = 144 B row stride (16-B aligned)

__device__ __forceinline__ void tdm_load_tile(unsigned int lds_off,
                                              const __bf16* gaddr,
                                              unsigned int tensor_d0,
                                              unsigned int tensor_d1,
                                              unsigned int tile_d0,
                                              unsigned int tile_d1) {
    const unsigned long long ga = (unsigned long long)(uintptr_t)gaddr;
    v4u g0;
    g0[0] = 1u;                                    // count=1, user descriptor
    g0[1] = lds_off;                               // lds_addr (bytes)
    g0[2] = (unsigned int)ga;                      // global_addr[31:0]
    g0[3] = (unsigned int)((ga >> 32) & 0x1FFFFFFu) | (2u << 30); // [56:32]|type=2

    v8i g1;
    g1[0] = (int)((1u << 16)        // data_size = 1 -> 2 bytes
                | (1u << 20)        // pad_enable
                | (4u << 22)        // pad_interval: 2^(4+1)=32 DWORDs (128 B)
                | (3u << 25));      // pad_amount: 4 DWORDs (16 B)
    g1[1] = (int)((tensor_d0 & 0xFFFFu) << 16);                    // dim0[15:0]
    g1[2] = (int)(((tensor_d0 >> 16) & 0xFFFFu)
                | ((tensor_d1 & 0xFFFFu) << 16));                  // dim0 hi | dim1 lo
    g1[3] = (int)(((tensor_d1 >> 16) & 0xFFFFu) | (tile_d0 << 16)); // dim1 hi | tile0
    g1[4] = (int)(tile_d1 & 0xFFFFu);                              // tile1 (tile2=0)
    g1[5] = (int)tensor_d0;                                        // dim0_stride lo
    g1[6] = 0;
    g1[7] = 0;

    const v4i z4 = {0, 0, 0, 0};
#if __clang_major__ >= 23
    const v8i z8 = {0, 0, 0, 0, 0, 0, 0, 0};
    __builtin_amdgcn_tensor_load_to_lds(g0, g1, z4, z4, z8, 0);
#else
    __builtin_amdgcn_tensor_load_to_lds(g0, g1, z4, z4, 0);
#endif
}

// ---------------- bf16 WMMA GEMM with TDM double-buffered staging ----------
//
// Block tile 128(M) x 128(N), K-step 64. 256 threads = 8 wave32 waves in
// 2(M) x 4(N); each wave computes 64x32 via 4x2 WMMA 16x16 tiles.
// Wave 0 drives the TDM: issue next tile pair, s_wait_tensorcnt<=2 for the
// current pair, workgroup barrier, compute.

__global__ __launch_bounds__(256)
void ternary_gemm_tdm_kernel(const __bf16* __restrict__ xq,
                             const __bf16* __restrict__ wq,
                             const float* __restrict__ bias,
                             float* __restrict__ out) {
    __shared__ __bf16 lds_x[2][BM][LDSK];
    __shared__ __bf16 lds_w[2][BN][LDSK];

    const int tid  = threadIdx.x;
    const int lane = tid & 31;
    const int wid  = tid >> 5;
    const int wm   = wid & 1;           // wave M index -> 64 rows
    const int wn   = wid >> 1;          // wave N index -> 32 cols

    const int m0 = blockIdx.y * BM;
    const int n0 = blockIdx.x * BN;

    const bool issuer = (wid == 0);

    unsigned int lx[2], lw[2];
    lx[0] = (unsigned int)(uintptr_t)&lds_x[0][0][0];
    lx[1] = (unsigned int)(uintptr_t)&lds_x[1][0][0];
    lw[0] = (unsigned int)(uintptr_t)&lds_w[0][0][0];
    lw[1] = (unsigned int)(uintptr_t)&lds_w[1][0][0];

    v8f acc[4][2];
#pragma unroll
    for (int mi = 0; mi < 4; ++mi)
#pragma unroll
        for (int ni = 0; ni < 2; ++ni)
            acc[mi][ni] = (v8f){0.f, 0.f, 0.f, 0.f, 0.f, 0.f, 0.f, 0.f};

    // Fragment addressing per ISA 16-bit A/B layout:
    //   lanes 0-15 : row = lane,    K chunks [0..7] and [16..23]
    //   lanes 16-31: row = lane-16, K chunks [8..15] and [24..31]
    const int frow  = lane & 15;
    const int fkoff = (lane >> 4) << 3;

    if (issuer) {
        tdm_load_tile(lx[0], xq + (size_t)m0 * D_IN, D_IN, TOKENS, BK, BM);
        tdm_load_tile(lw[0], wq + (size_t)n0 * D_IN, D_IN, D_OUT, BK, BN);
    }

    int cur = 0;
    for (int kt = 0; kt < D_IN; kt += BK) {
        if (issuer) {
            if (kt + BK < D_IN) {
                tdm_load_tile(lx[cur ^ 1], xq + (size_t)m0 * D_IN + kt + BK,
                              D_IN, TOKENS, BK, BM);
                tdm_load_tile(lw[cur ^ 1], wq + (size_t)n0 * D_IN + kt + BK,
                              D_IN, D_OUT, BK, BN);
                __builtin_amdgcn_s_wait_tensorcnt(2);  // current pair done
            } else {
                __builtin_amdgcn_s_wait_tensorcnt(0);
            }
        }
        __syncthreads();

#pragma unroll
        for (int kk = 0; kk < BK; kk += 32) {
            const int kb = kk + fkoff;

            v16bf afrag[4];
#pragma unroll
            for (int mi = 0; mi < 4; ++mi) {
                const int r = wm * 64 + mi * 16 + frow;
                v8bf c0 = *(const v8bf*)&lds_x[cur][r][kb];
                v8bf c1 = *(const v8bf*)&lds_x[cur][r][kb + 16];
                afrag[mi] = __builtin_shufflevector(c0, c1,
                    0, 1, 2, 3, 4, 5, 6, 7, 8, 9, 10, 11, 12, 13, 14, 15);
            }

            v16bf bfrag[2];
#pragma unroll
            for (int ni = 0; ni < 2; ++ni) {
                const int r = wn * 32 + ni * 16 + frow;
                v8bf c0 = *(const v8bf*)&lds_w[cur][r][kb];
                v8bf c1 = *(const v8bf*)&lds_w[cur][r][kb + 16];
                bfrag[ni] = __builtin_shufflevector(c0, c1,
                    0, 1, 2, 3, 4, 5, 6, 7, 8, 9, 10, 11, 12, 13, 14, 15);
            }

#pragma unroll
            for (int mi = 0; mi < 4; ++mi)
#pragma unroll
                for (int ni = 0; ni < 2; ++ni)
                    acc[mi][ni] = __builtin_amdgcn_wmma_f32_16x16x32_bf16(
                        false, afrag[mi], false, bfrag[ni],
                        (short)0, acc[mi][ni], false, false);
        }

        __syncthreads();   // all waves done reading buf before TDM overwrites
        cur ^= 1;
    }

    // Epilogue: C/D layout -> VGPR r holds M=r (lanes 0-15) / M=r+8 (16-31),
    // N = lane%16. Add bias, store f32.
    const int mhi = (lane >> 4) << 3;
#pragma unroll
    for (int ni = 0; ni < 2; ++ni) {
        const int ncol = n0 + wn * 32 + ni * 16 + frow;
        const float bv = bias[ncol];
#pragma unroll
        for (int mi = 0; mi < 4; ++mi) {
            const int mbase = m0 + wm * 64 + mi * 16 + mhi;
#pragma unroll
            for (int r = 0; r < 8; ++r) {
                out[(size_t)(mbase + r) * D_OUT + ncol] = acc[mi][ni][r] + bv;
            }
        }
    }
}

// ---------------- fallback: fused quantize-in-staging GEMM (round 1) -------

__global__ __launch_bounds__(256)
void ternary_gemm_fused_kernel(const float* __restrict__ x,
                               const float* __restrict__ w,
                               const float* __restrict__ bias,
                               const float* __restrict__ alpha_ptr,
                               float* __restrict__ out) {
    __shared__ __bf16 lds_x[BM][LDSK];
    __shared__ __bf16 lds_w[BN][LDSK];

    const int tid  = threadIdx.x;
    const int lane = tid & 31;
    const int wid  = tid >> 5;
    const int wm   = wid & 1;
    const int wn   = wid >> 1;

    const int m0 = blockIdx.y * BM;
    const int n0 = blockIdx.x * BN;

    const float thr = 0.5f * alpha_ptr[0];

    v8f acc[4][2];
#pragma unroll
    for (int mi = 0; mi < 4; ++mi)
#pragma unroll
        for (int ni = 0; ni < 2; ++ni)
            acc[mi][ni] = (v8f){0.f, 0.f, 0.f, 0.f, 0.f, 0.f, 0.f, 0.f};

    const int frow  = lane & 15;
    const int fkoff = (lane >> 4) << 3;

    for (int kt = 0; kt < D_IN; kt += BK) {
#pragma unroll
        for (int i = 0; i < 8; ++i) {
            const int f   = i * 256 + tid;
            const int row = f >> 4;
            const int col = (f & 15) << 2;
            const float4 vx = *(const float4*)(x + (size_t)(m0 + row) * D_IN + kt + col);
            v4bf px;
            px[0] = (__bf16)vx.x; px[1] = (__bf16)vx.y;
            px[2] = (__bf16)vx.z; px[3] = (__bf16)vx.w;
            *(v4bf*)&lds_x[row][col] = px;

            const float4 vw = *(const float4*)(w + (size_t)(n0 + row) * D_IN + kt + col);
            v4bf pw;
            pw[0] = (__bf16)((fabsf(vw.x) > thr) ? (vw.x > 0.f ? 1.f : -1.f) : 0.f);
            pw[1] = (__bf16)((fabsf(vw.y) > thr) ? (vw.y > 0.f ? 1.f : -1.f) : 0.f);
            pw[2] = (__bf16)((fabsf(vw.z) > thr) ? (vw.z > 0.f ? 1.f : -1.f) : 0.f);
            pw[3] = (__bf16)((fabsf(vw.w) > thr) ? (vw.w > 0.f ? 1.f : -1.f) : 0.f);
            *(v4bf*)&lds_w[row][col] = pw;
        }

        __syncthreads();

#pragma unroll
        for (int kk = 0; kk < BK; kk += 32) {
            const int kb = kk + fkoff;
            v16bf afrag[4];
#pragma unroll
            for (int mi = 0; mi < 4; ++mi) {
                const int r = wm * 64 + mi * 16 + frow;
                v8bf c0 = *(const v8bf*)&lds_x[r][kb];
                v8bf c1 = *(const v8bf*)&lds_x[r][kb + 16];
                afrag[mi] = __builtin_shufflevector(c0, c1,
                    0, 1, 2, 3, 4, 5, 6, 7, 8, 9, 10, 11, 12, 13, 14, 15);
            }
            v16bf bfrag[2];
#pragma unroll
            for (int ni = 0; ni < 2; ++ni) {
                const int r = wn * 32 + ni * 16 + frow;
                v8bf c0 = *(const v8bf*)&lds_w[r][kb];
                v8bf c1 = *(const v8bf*)&lds_w[r][kb + 16];
                bfrag[ni] = __builtin_shufflevector(c0, c1,
                    0, 1, 2, 3, 4, 5, 6, 7, 8, 9, 10, 11, 12, 13, 14, 15);
            }
#pragma unroll
            for (int mi = 0; mi < 4; ++mi)
#pragma unroll
                for (int ni = 0; ni < 2; ++ni)
                    acc[mi][ni] = __builtin_amdgcn_wmma_f32_16x16x32_bf16(
                        false, afrag[mi], false, bfrag[ni],
                        (short)0, acc[mi][ni], false, false);
        }

        __syncthreads();
    }

    const int mhi = (lane >> 4) << 3;
#pragma unroll
    for (int ni = 0; ni < 2; ++ni) {
        const int ncol = n0 + wn * 32 + ni * 16 + frow;
        const float bv = bias[ncol];
#pragma unroll
        for (int mi = 0; mi < 4; ++mi) {
            const int mbase = m0 + wm * 64 + mi * 16 + mhi;
#pragma unroll
            for (int r = 0; r < 8; ++r) {
                out[(size_t)(mbase + r) * D_OUT + ncol] = acc[mi][ni][r] + bv;
            }
        }
    }
}

// ---------------------------------------------------------------------------

extern "C" void kernel_launch(void* const* d_in, const int* in_sizes, int n_in,
                              void* d_out, int out_size, void* d_ws, size_t ws_size,
                              hipStream_t stream) {
    const float* x    = (const float*)d_in[0];   // TOKENS x D_IN
    const float* w    = (const float*)d_in[1];   // D_OUT x D_IN
    const float* bias = (const float*)d_in[2];   // D_OUT
    float*       out  = (float*)d_out;           // TOKENS x D_OUT

    float* partial = (float*)d_ws;               // 1024 floats
    float* alpha   = partial + 1024;             // 1 float

    const int wn = in_sizes[1];                  // D_OUT * D_IN

    absmean_partial_kernel<<<1024, 256, 0, stream>>>(w, partial, wn);
    absmean_final_kernel<<<1, 256, 0, stream>>>(partial, alpha, 1024,
                                                1.0f / (float)wn);

    dim3 grid(D_OUT / BN, TOKENS / BM);

    const size_t xq_bytes = (size_t)2 * TOKENS * D_IN;   // 64 MB
    const size_t wq_bytes = (size_t)2 * D_OUT * D_IN;    // 128 MB
    const size_t need = 8192 + xq_bytes + wq_bytes;

    if (ws_size >= need) {
        __bf16* xq = (__bf16*)((char*)d_ws + 8192);
        __bf16* wq = (__bf16*)((char*)d_ws + 8192 + xq_bytes);

        quant_w_kernel<<<4096, 256, 0, stream>>>(w, alpha, wq, wn / 4);
        conv_x_kernel<<<4096, 256, 0, stream>>>(x, xq, in_sizes[0] / 4);
        ternary_gemm_tdm_kernel<<<grid, 256, 0, stream>>>(xq, wq, bias, out);
    } else {
        ternary_gemm_fused_kernel<<<grid, 256, 0, stream>>>(x, w, bias, alpha, out);
    }
}
